// STransformer_54546084659401
// MI455X (gfx1250) — matrix operand order repaired and checked by reading, not compiled, not run on hardware
//
#include <hip/hip_runtime.h>

// ---------------------------------------------------------------------------
// CDNA5 / gfx1250 implementation of the STransformer reference.
// All dense contractions go through v_wmma_f32_16x16x32_bf16 (f32 accum).
// Attention is fused (flash-style online softmax) so the 384x400x400 energy
// tensor never hits memory.  All hot loops are branchless: clamped-address
// float2 loads + v_cndmask selects feed the WMMA fragments.
// ---------------------------------------------------------------------------

typedef __attribute__((ext_vector_type(16))) __bf16 v16bf;
typedef __attribute__((ext_vector_type(8)))  float  v8f;

#define DEV_INLINE __device__ __forceinline__

namespace stx {
constexpr int cB = 16, cC = 32, cN = 400, cT = 12;
constexpr int cR = 64, cG = 16, cE = 64, cDH = 32, cFE = 256;
constexpr long long TOK = (long long)cB * cN * cT;   // 76800 tokens
}

DEV_INLINE __bf16 to_bf16(float x) { return (__bf16)x; }

// A-operand (16x32 bf16): lanes hold row M=lane%16; lo lane-half holds
// K=0..15, hi half K=16..31 (same k order as B).
DEV_INLINE int kmapA(int e, int lhi) { return e + 16 * lhi; }
// B-operand (32x16 bf16): lanes 0-15 hold K=0..15, lanes 16-31 hold K=16..31.
DEV_INLINE int kmapB(int e, int lhi) { return e + 16 * lhi; }

DEV_INLINE v8f wmma_bf16(v16bf a, v16bf b, v8f c) {
  return __builtin_amdgcn_wmma_f32_16x16x32_bf16(false, a, false, b,
                                                 (short)0, c, false, false);
}

// ---------------------------------------------------------------------------
// Generic batched 16x16-tile WMMA GEMM (one wave per tile).
//   D = act2( beta*Cin + alpha*act1( A*B + bias ) )
// A element (m,k): TA ? A[k*lda+m] : A[m*lda+k]
// B element (k,n): TB ? B[n*ldb+k] : B[k*ldb+n]
// store: storeT ? D[n*ldd+m] : D[m*ldd+n]
// biasMode: 0 none, 1 per-m, 2 per-n.  act: 0 none, 1 relu, 2 sigmoid.
// Requirements (all satisfied by this model): K even, lda/ldb even for the
// float2 paths, base pointers 8B aligned.
// ---------------------------------------------------------------------------
template <int TA, int TB>
__global__ __launch_bounds__(32)
void gemm16_kernel(const float* __restrict__ A, const float* __restrict__ B,
                   const float* __restrict__ Cin, const float* __restrict__ bias,
                   float* __restrict__ D,
                   int M, int N, int K,
                   int lda, int ldb, int ldc, int ldd,
                   long long sA, long long sB, long long sC, long long sD,
                   int biasMode, int act1, int act2,
                   float alpha, float beta, int storeT, int tilesN)
{
  const int lane = threadIdx.x;
  const int l15 = lane & 15, lhi = lane >> 4;
  const int tile = blockIdx.x;
  const int tm = (tile / tilesN) * 16;
  const int tn = (tile % tilesN) * 16;
  const int bidx = blockIdx.y;

  A += sA * bidx;
  B += sB * bidx;
  D += sD * bidx;
  if (Cin) Cin += sC * bidx;

  const int ma = tm + l15;                 // A fragment row
  const int nb = tn + l15;                 // B fragment col
  const bool maok = ma < M;
  const bool nbok = nb < N;
  const int mac = maok ? ma : 0;           // clamped (always-legal) indices
  const int nbc = nbok ? nb : 0;
  const long long arow = (long long)mac * lda;   // row base for TA==0
  const long long brow = (long long)nbc * ldb;   // row base for TB==1

  v8f acc = {};
  for (int k0 = 0; k0 < K; k0 += 32) {
    if (TA == 0 && k0 + 32 < K) __builtin_prefetch(A + arow + k0 + 32, 0, 0);
    if (TB == 1 && k0 + 32 < K) __builtin_prefetch(B + brow + k0 + 32, 0, 0);
    v16bf af, bf;
#pragma unroll
    for (int j = 0; j < 8; ++j) {
      // ---- A pair (k, k+1) ----
      const int ka = k0 + kmapA(2 * j, lhi);
      const bool aok = maok && (ka < K);         // K even -> ka+1 < K too
      float a0, a1;
      if (TA == 0) {
        const float2 p = *(const float2*)(A + arow + (aok ? ka : 0));
        a0 = aok ? p.x : 0.f;
        a1 = aok ? p.y : 0.f;
      } else {
        const long long i0 = (long long)(aok ? ka : 0) * lda + mac;
        const float q0 = A[i0];
        const float q1 = A[i0 + (aok ? lda : 0)];
        a0 = aok ? q0 : 0.f;
        a1 = aok ? q1 : 0.f;
      }
      af[2 * j]     = to_bf16(a0);
      af[2 * j + 1] = to_bf16(a1);
      // ---- B pair (k, k+1) ----
      const int kb = k0 + kmapB(2 * j, lhi);
      const bool bok = nbok && (kb < K);
      float b0, b1;
      if (TB == 1) {
        const float2 p = *(const float2*)(B + brow + (bok ? kb : 0));
        b0 = bok ? p.x : 0.f;
        b1 = bok ? p.y : 0.f;
      } else {
        const long long i0 = (long long)(bok ? kb : 0) * ldb + nbc;
        const float q0 = B[i0];
        const float q1 = B[i0 + (bok ? ldb : 0)];
        b0 = bok ? q0 : 0.f;
        b1 = bok ? q1 : 0.f;
      }
      bf[2 * j]     = to_bf16(b0);
      bf[2 * j + 1] = to_bf16(b1);
    }
    acc = wmma_bf16(af, bf, acc);
  }

  // C/D layout: lane l15 holds col n; VGPR v holds row v + 8*lhi.
  const int n = tn + l15;
#pragma unroll
  for (int v = 0; v < 8; ++v) {
    const int m = tm + v + 8 * lhi;
    if (m < M && n < N) {
      float r = acc[v];
      if (biasMode == 1) r += bias[m];
      else if (biasMode == 2) r += bias[n];
      if (act1 == 1) r = fmaxf(r, 0.f);
      r *= alpha;
      if (beta != 0.f) r += beta * Cin[(long long)m * ldc + n];
      if (act2 == 1) r = fmaxf(r, 0.f);
      else if (act2 == 2) r = 1.f / (1.f + __expf(-r));
      if (storeT) D[(long long)n * ldd + m] = r;
      else        D[(long long)m * ldd + n] = r;
    }
  }
}

// ---------------------------------------------------------------------------
// Fused attention: per (b,t,h, 16-row q tile), online softmax over 400 keys.
// qq/kk/vv/out layout: (B, N, T, H*DH) f32, row (n) stride = T*E = 768.
// N = 400 is a multiple of 16 -> Q fragment and output stores need no guards.
// ---------------------------------------------------------------------------
__global__ __launch_bounds__(32)
void attn_kernel(const float* __restrict__ qq, const float* __restrict__ kk,
                 const float* __restrict__ vv, float* __restrict__ out)
{
  using namespace stx;
  const int lane = threadIdx.x;
  const int l15 = lane & 15, lhi = lane >> 4;
  const int q0 = blockIdx.x * 16;
  const int th = blockIdx.y;               // t*H + h
  const int t = th >> 1, h = th & 1;
  const int b = blockIdx.z;
  const long long base = (long long)b * cN * cT * cE + (long long)t * cE + h * cDH;
  const int rstride = cT * cE;             // 768

  __shared__ float sS[16][34];             // 34: keeps float2 LDS reads aligned
  __shared__ float sM[16], sL[16], sC[16];

  // Q fragment (A-operand, K = DH = 32): row = q0 + l15 (always valid)
  v16bf qf;
  {
    const float* qrow = qq + base + (long long)(q0 + l15) * rstride;
#pragma unroll
    for (int j = 0; j < 8; ++j) {
      const float2 p = *(const float2*)(qrow + kmapA(2 * j, lhi));
      qf[2 * j]     = to_bf16(p.x);
      qf[2 * j + 1] = to_bf16(p.y);
    }
  }
  if (lane < 16) { sM[lane] = -1e30f; sL[lane] = 0.f; }

  v8f o0 = {}, o1 = {};
  const float scale = 0.125f;              // 1/sqrt(E), E = 64

  for (int kb = 0; kb < cN; kb += 32) {
    // K^T fragments (B-operand 32x16): col = key index, K-dim = d.
    // kc0 <= 399 is always valid; kc1 needs a clamp in the final iteration.
    v16bf kf0, kf1;
    const int kc1 = kb + 16 + l15;
    const bool k1ok = kc1 < cN;
    {
      const float* k0row = kk + base + (long long)(kb + l15) * rstride;
      const float* k1row = kk + base + (long long)(k1ok ? kc1 : 0) * rstride;
#pragma unroll
      for (int j = 0; j < 8; ++j) {
        const int d = kmapB(2 * j, lhi);
        const float2 p0 = *(const float2*)(k0row + d);
        const float2 p1 = *(const float2*)(k1row + d);
        kf0[2 * j]     = to_bf16(p0.x);
        kf0[2 * j + 1] = to_bf16(p0.y);
        kf1[2 * j]     = to_bf16(k1ok ? p1.x : 0.f);
        kf1[2 * j + 1] = to_bf16(k1ok ? p1.y : 0.f);
      }
    }
    v8f s0 = {}, s1 = {};
    s0 = wmma_bf16(qf, kf0, s0);
    s1 = wmma_bf16(qf, kf1, s1);

    __syncthreads();                       // sS reads of prev iteration done
#pragma unroll
    for (int v = 0; v < 8; ++v) {
      const int m = v + 8 * lhi;
      sS[m][l15]      = s0[v] * scale;                         // always valid
      sS[m][16 + l15] = k1ok ? s1[v] * scale : -1e30f;         // mask pad keys
    }
    __syncthreads();

    if (lane < 16) {                       // one lane per q-row: online softmax
      const int r = lane;
      const float mo = sM[r], lo = sL[r];
      float mx = mo;
      for (int j = 0; j < 32; ++j) mx = fmaxf(mx, sS[r][j]);
      const float corr = __expf(mo - mx);
      float ls = 0.f;
      for (int j = 0; j < 32; ++j) {
        const float p = __expf(sS[r][j] - mx);
        sS[r][j] = p;
        ls += p;
      }
      sM[r] = mx; sL[r] = lo * corr + ls; sC[r] = corr;
    }
    __syncthreads();

#pragma unroll
    for (int v = 0; v < 8; ++v) {          // rescale running output
      const float cr = sC[v + 8 * lhi];
      o0[v] *= cr; o1[v] *= cr;
    }

    v16bf pf;                              // P as A-operand (LDS did the shuffle)
#pragma unroll
    for (int j = 0; j < 8; ++j) {
      const float2 p = *(const float2*)(&sS[l15][kmapA(2 * j, lhi)]);
      pf[2 * j]     = to_bf16(p.x);
      pf[2 * j + 1] = to_bf16(p.y);
    }

    v16bf vf0, vf1;                        // V as B-operand, two 16-wide d slices
#pragma unroll
    for (int e = 0; e < 16; ++e) {
      const int kg = kb + kmapB(e, lhi);
      const bool ok = kg < cN;
      const long long vr = base + (long long)(ok ? kg : 0) * rstride;
      const float v0 = vv[vr + l15];
      const float v1 = vv[vr + 16 + l15];
      vf0[e] = to_bf16(ok ? v0 : 0.f);
      vf1[e] = to_bf16(ok ? v1 : 0.f);
    }
    o0 = wmma_bf16(pf, vf0, o0);
    o1 = wmma_bf16(pf, vf1, o1);
  }

  __syncthreads();
#pragma unroll
  for (int v = 0; v < 8; ++v) {
    const int m = v + 8 * lhi;
    const float inv = 1.f / sL[m];
    const long long orow = base + (long long)(q0 + m) * rstride;
    out[orow + l15]      = o0[v] * inv;
    out[orow + 16 + l15] = o1[v] * inv;
  }
}

// ---------------------------------------------------------------------------
// LayerNorm over E=64: one row per 64-thread block.  out = LN(a + r)*w + b
// ---------------------------------------------------------------------------
__global__ __launch_bounds__(64)
void ln_residual_kernel(const float* __restrict__ a, const float* __restrict__ r,
                        const float* __restrict__ w, const float* __restrict__ bb,
                        float* __restrict__ out)
{
  const long long row = blockIdx.x;
  const int e = threadIdx.x;
  __shared__ float red[64];
  const float v = a[row * 64 + e] + r[row * 64 + e];
  red[e] = v;
  __syncthreads();
  for (int s = 32; s > 0; s >>= 1) { if (e < s) red[e] += red[e + s]; __syncthreads(); }
  const float mean = red[0] * (1.f / 64.f);
  __syncthreads();
  const float d = v - mean;
  red[e] = d * d;
  __syncthreads();
  for (int s = 32; s > 0; s >>= 1) { if (e < s) red[e] += red[e + s]; __syncthreads(); }
  const float var = red[0] * (1.f / 64.f);
  out[row * 64 + e] = d * rsqrtf(var + 1e-5f) * w[e] + bb[e];
}

// q[b,n,t,:] += ds[n,:]
__global__ void q_add_ds_kernel(float* __restrict__ q, const float* __restrict__ ds,
                                long long total)
{
  for (long long i = blockIdx.x * (long long)blockDim.x + threadIdx.x; i < total;
       i += (long long)gridDim.x * blockDim.x) {
    const int e = (int)(i & 63);
    long long rr = i >> 6;           // (b,n,t)
    rr /= stx::cT;                   // (b,n)
    const int n = (int)(rr % stx::cN);
    q[i] += ds[n * 64 + e];
  }
}

// out = g*u + (1-g)*xc
__global__ void gate_mix_kernel(const float* __restrict__ g, const float* __restrict__ u,
                                const float* __restrict__ xc, float* __restrict__ out,
                                long long total)
{
  for (long long i = blockIdx.x * (long long)blockDim.x + threadIdx.x; i < total;
       i += (long long)gridDim.x * blockDim.x) {
    const float gv = g[i];
    out[i] = gv * u[i] + (1.f - gv) * xc[i];
  }
}

// Ag = relu(xg_mat @ xgt_mat - 0.5) with the reference's scrambled flattenings.
__global__ __launch_bounds__(256)
void ag_kernel(const float* __restrict__ xs, float* __restrict__ Ag)
{
  using namespace stx;
  const int tid = threadIdx.x;           // 256 = 16*16
  const int g1 = tid >> 4, g2 = tid & 15;
  const int NI = cC * cB * cT;           // 6144
  float s = 0.f;
  for (int i = 0; i < NI; ++i) {
    const int c = i / (cB * cT);
    const int rem = i - c * (cB * cT);
    const int b = rem / cT;
    const int t = rem - b * cT;
    const int t2 = i / (cB * cC);
    const int rem2 = i - t2 * (cB * cC);
    const int b2 = rem2 / cC;
    const int c2 = rem2 - b2 * cC;
    const float xa = xs[((b * cC + c) * cG + g1) * cT + t];
    const float xb = xs[((b2 * cC + c2) * cG + g2) * cT + t2];
    s += xa * xb;
  }
  const float v = s - 0.5f;
  Ag[tid] = v > 0.f ? v : 0.f;
}

// sg0 = softmax(asym_adj(Ag)); sg1 = softmax(asym_adj(Ag^T))  (16x16)
__global__ __launch_bounds__(256)
void sg_kernel(const float* __restrict__ Ag, float* __restrict__ sg0,
               float* __restrict__ sg1)
{
  __shared__ float A[16][16];
  __shared__ float rinv[16], cinv[16];
  const int tid = threadIdx.x;
  const int r = tid >> 4, c = tid & 15;
  A[r][c] = Ag[tid];
  __syncthreads();
  if (tid < 16) {
    float s = 0.f, s2 = 0.f;
    for (int j = 0; j < 16; ++j) { s += A[tid][j]; s2 += A[j][tid]; }
    rinv[tid] = (s  > 0.f) ? 1.f / s  : 0.f;
    cinv[tid] = (s2 > 0.f) ? 1.f / s2 : 0.f;
  }
  __syncthreads();
  if (tid < 16) {
    const int rr = tid;
    float mx = -1e30f, se = 0.f;
    for (int j = 0; j < 16; ++j) mx = fmaxf(mx, rinv[rr] * A[rr][j]);
    for (int j = 0; j < 16; ++j) se += __expf(rinv[rr] * A[rr][j] - mx);
    for (int j = 0; j < 16; ++j) sg0[rr * 16 + j] = __expf(rinv[rr] * A[rr][j] - mx) / se;
    mx = -1e30f; se = 0.f;
    for (int j = 0; j < 16; ++j) mx = fmaxf(mx, cinv[rr] * A[j][rr]);
    for (int j = 0; j < 16; ++j) se += __expf(cinv[rr] * A[j][rr] - mx);
    for (int j = 0; j < 16; ++j) sg1[rr * 16 + j] = __expf(cinv[rr] * A[j][rr] - mx) / se;
  }
}

// ---------------------------------------------------------------------------
// Host side
// ---------------------------------------------------------------------------
static inline void gemm(hipStream_t st,
                        const float* A, int lda, long long sA, int tA,
                        const float* Bm, int ldb, long long sB, int tB,
                        const float* Cin, int ldc, long long sC,
                        float* D, int ldd, long long sD, int storeT,
                        const float* bias, int biasMode,
                        int M, int N, int K, int batch,
                        float alpha, float beta, int act1, int act2)
{
  const int tilesM = (M + 15) / 16, tilesN = (N + 15) / 16;
  dim3 grid((unsigned)(tilesM * tilesN), (unsigned)batch);
#define GEMM_ARGS A, Bm, Cin, bias, D, M, N, K, lda, ldb, ldc, ldd, \
                  sA, sB, sC, sD, biasMode, act1, act2, alpha, beta, storeT, tilesN
  if (tA == 0 && tB == 0)      gemm16_kernel<0, 0><<<grid, 32, 0, st>>>(GEMM_ARGS);
  else if (tA == 0 && tB == 1) gemm16_kernel<0, 1><<<grid, 32, 0, st>>>(GEMM_ARGS);
  else if (tA == 1 && tB == 0) gemm16_kernel<1, 0><<<grid, 32, 0, st>>>(GEMM_ARGS);
  else                         gemm16_kernel<1, 1><<<grid, 32, 0, st>>>(GEMM_ARGS);
#undef GEMM_ARGS
}

extern "C" void kernel_launch(void* const* d_in, const int* in_sizes, int n_in,
                              void* d_out, int out_size, void* d_ws, size_t ws_size,
                              hipStream_t stream)
{
  (void)in_sizes; (void)n_in; (void)out_size;
  using namespace stx;

  const float* query = (const float*)d_in[2];
  const float* sup0  = (const float*)d_in[3];
  const float* sup1  = (const float*)d_in[4];
  const float* supr0 = (const float*)d_in[5];
  const float* supr1 = (const float*)d_in[6];
  const float* Mrg   = (const float*)d_in[7];   // (R,G)
  const float* Mor   = (const float*)d_in[8];   // (N,R)
  const float* D_S   = (const float*)d_in[9];   // (N,N)
  const float* Wc1   = (const float*)d_in[10];  // (E,C)
  const float* bc1   = (const float*)d_in[11];
  const float* Wc11  = (const float*)d_in[12];  // (C,E)
  const float* bc11  = (const float*)d_in[13];
  const float* Wg    = (const float*)d_in[14];  // (C,3C)
  const float* bg    = (const float*)d_in[15];
  const float* Wemb  = (const float*)d_in[16];  // (E,N)
  const float* bemb  = (const float*)d_in[17];
  const float* Wq    = (const float*)d_in[18];
  const float* Wk    = (const float*)d_in[19];
  const float* Wv    = (const float*)d_in[20];
  const float* Wfc   = (const float*)d_in[21];
  const float* bfc   = (const float*)d_in[22];
  const float* ln1w  = (const float*)d_in[23];
  const float* ln1b  = (const float*)d_in[24];
  const float* ln2w  = (const float*)d_in[25];
  const float* ln2b  = (const float*)d_in[26];
  const float* Wff1  = (const float*)d_in[27];  // (FE,E)
  const float* bff1  = (const float*)d_in[28];
  const float* Wff2  = (const float*)d_in[29];  // (E,FE)
  const float* bff2  = (const float*)d_in[30];
  const float* Wfs   = (const float*)d_in[31];
  const float* bfs   = (const float*)d_in[32];
  const float* Wfg   = (const float*)d_in[33];
  const float* bfg   = (const float*)d_in[34];

  // ---- workspace arena (floats) ----
  float* ws = (float*)d_ws;
  size_t off = 0;
  auto alloc = [&](size_t n) { float* p = ws + off; off += n; return p; };
  const size_t BCNT = (size_t)cB * cC * cN * cT;   // 2,457,600
  const size_t BCRT = (size_t)cB * cC * cR * cT;   //   393,216
  const size_t BCGT = (size_t)cB * cC * cG * cT;   //    98,304
  const size_t BNTE = (size_t)cB * cN * cT * cE;   // 4,915,200

  float* xs0   = alloc(BCNT);
  float* xs1   = alloc(BCNT);
  float* ho    = alloc(BCNT);
  float* x_r   = alloc(BCRT);
  float* xr0   = alloc(BCRT);
  float* xr1   = alloc(BCRT);
  float* hr    = alloc(BCRT);
  float* xsb   = alloc(BCGT);
  float* xsg0  = alloc(BCGT);
  float* xsg1  = alloc(BCGT);
  float* xg    = alloc(BCGT);
  float* Agb   = alloc(256);
  float* sg0   = alloc(256);
  float* sg1   = alloc(256);
  float* dsb   = alloc((size_t)cN * cE);
  float* Xc    = alloc(BNTE);
  float* qb    = alloc(BNTE);
  float* qqb   = alloc(BNTE);
  float* kkb   = alloc(BNTE);
  float* vvb   = alloc(BNTE);
  float* ato   = alloc(BNTE);     // attn out -> later reused as ff2
  float* afc   = alloc(BNTE);     // attn fc  -> later reused as U_S
  float* xln   = alloc(BNTE);
  float* ff1   = alloc((size_t)cB * cN * cT * cFE);
  if (off * sizeof(float) > ws_size) return;   // workspace too small: bail

  float* out_main = (float*)d_out;                          // (B,C,N,T)
  float* out_xc   = out_main + BCNT;                        // (B,C,R,T)
  float* out_xs   = out_main + BCNT + BCRT;                 // (B,C,G,T)

  const int BC = cB * cC;                                   // 512
  const long long sNT = (long long)cN * cT;                 // 4800
  const long long sRT = (long long)cR * cT;                 // 768
  const long long sGT = (long long)cG * cT;                 // 192
  const long long sBNT = (long long)cC * cN * cT;           // per-b (C,N,T)

  // ===== _chgcn =====
  // xs0 = x @ s0, xs1 = x @ s1 (per (b,c): s^T (NxN) @ x (NxT))
  gemm(stream, sup0, cN, 0, 1, query, cT, sNT, 0, nullptr, 0, 0,
       xs0, cT, sNT, 0, nullptr, 0, cN, cT, cN, BC, 1.f, 0.f, 0, 0);
  gemm(stream, sup1, cN, 0, 1, query, cT, sNT, 0, nullptr, 0, 0,
       xs1, cT, sNT, 0, nullptr, 0, cN, cT, cN, BC, 1.f, 0.f, 0, 0);
  // ho = Wg[:, :C]@x + Wg[:,C:2C]@xs0 + Wg[:,2C:]@xs1 + bg  (per b: 32x4800)
  gemm(stream, Wg, 3 * cC, 0, 0, query, (int)sNT, sBNT, 0, nullptr, 0, 0,
       ho, (int)sNT, sBNT, 0, bg, 1, cC, (int)sNT, cC, cB, 1.f, 0.f, 0, 0);
  gemm(stream, Wg + cC, 3 * cC, 0, 0, xs0, (int)sNT, sBNT, 0, ho, (int)sNT, sBNT,
       ho, (int)sNT, sBNT, 0, nullptr, 0, cC, (int)sNT, cC, cB, 1.f, 1.f, 0, 0);
  gemm(stream, Wg + 2 * cC, 3 * cC, 0, 0, xs1, (int)sNT, sBNT, 0, ho, (int)sNT, sBNT,
       ho, (int)sNT, sBNT, 0, nullptr, 0, cC, (int)sNT, cC, cB, 1.f, 1.f, 0, 0);

  // x_r = Mor^T @ x (per (b,c): 64xT)
  gemm(stream, Mor, cR, 0, 1, query, cT, sNT, 0, nullptr, 0, 0,
       x_r, cT, sRT, 0, nullptr, 0, cR, cT, cN, BC, 1.f, 0.f, 0, 0);
  // xr0/xr1 = x_r @ supr{0,1}
  gemm(stream, supr0, cR, 0, 1, x_r, cT, sRT, 0, nullptr, 0, 0,
       xr0, cT, sRT, 0, nullptr, 0, cR, cT, cR, BC, 1.f, 0.f, 0, 0);
  gemm(stream, supr1, cR, 0, 1, x_r, cT, sRT, 0, nullptr, 0, 0,
       xr1, cT, sRT, 0, nullptr, 0, cR, cT, cR, BC, 1.f, 0.f, 0, 0);
  // hr = gcn conv (per b: 32 x 768)
  const long long sBRT = (long long)cC * cR * cT;           // 24576
  gemm(stream, Wg, 3 * cC, 0, 0, x_r, (int)sRT, sBRT, 0, nullptr, 0, 0,
       hr, (int)sRT, sBRT, 0, bg, 1, cC, (int)sRT, cC, cB, 1.f, 0.f, 0, 0);
  gemm(stream, Wg + cC, 3 * cC, 0, 0, xr0, (int)sRT, sBRT, 0, hr, (int)sRT, sBRT,
       hr, (int)sRT, sBRT, 0, nullptr, 0, cC, (int)sRT, cC, cB, 1.f, 1.f, 0, 0);
  gemm(stream, Wg + 2 * cC, 3 * cC, 0, 0, xr1, (int)sRT, sBRT, 0, hr, (int)sRT, sBRT,
       hr, (int)sRT, sBRT, 0, nullptr, 0, cC, (int)sRT, cC, cB, 1.f, 1.f, 0, 0);

  // xs = Mrg^T @ x_r (per (b,c): 16xT)
  gemm(stream, Mrg, cG, 0, 1, x_r, cT, sRT, 0, nullptr, 0, 0,
       xsb, cT, sGT, 0, nullptr, 0, cG, cT, cR, BC, 1.f, 0.f, 0, 0);

  // data-dependent 16x16 adjacency + its two softmaxed asym_adj variants
  ag_kernel<<<1, 256, 0, stream>>>(xsb, Agb);
  sg_kernel<<<1, 256, 0, stream>>>(Agb, sg0, sg1);

  // xsg0/xsg1 = xs @ sg{0,1}
  gemm(stream, sg0, cG, 0, 1, xsb, cT, sGT, 0, nullptr, 0, 0,
       xsg0, cT, sGT, 0, nullptr, 0, cG, cT, cG, BC, 1.f, 0.f, 0, 0);
  gemm(stream, sg1, cG, 0, 1, xsb, cT, sGT, 0, nullptr, 0, 0,
       xsg1, cT, sGT, 0, nullptr, 0, cG, cT, cG, BC, 1.f, 0.f, 0, 0);
  // xg = gcn conv (per b: 32 x 192)
  const long long sBGT = (long long)cC * cG * cT;           // 6144
  gemm(stream, Wg, 3 * cC, 0, 0, xsb, (int)sGT, sBGT, 0, nullptr, 0, 0,
       xg, (int)sGT, sBGT, 0, bg, 1, cC, (int)sGT, cC, cB, 1.f, 0.f, 0, 0);
  gemm(stream, Wg + cC, 3 * cC, 0, 0, xsg0, (int)sGT, sBGT, 0, xg, (int)sGT, sBGT,
       xg, (int)sGT, sBGT, 0, nullptr, 0, cC, (int)sGT, cC, cB, 1.f, 1.f, 0, 0);
  gemm(stream, Wg + 2 * cC, 3 * cC, 0, 0, xsg1, (int)sGT, sBGT, 0, xg, (int)sGT, sBGT,
       xg, (int)sGT, sBGT, 0, nullptr, 0, cC, (int)sGT, cC, cB, 1.f, 1.f, 0, 0);

  // hr += 0.5*relu(Mrg @ xg); ho += 0.5*relu(Mor @ hr);
  // hr_final = hr + 0.5*relu(Mor^T @ ho) -> out_xc;
  // xg_final = xg + 0.5*relu(Mrg^T @ hr_final) -> out_xs
  gemm(stream, Mrg, cG, 0, 0, xg, cT, sGT, 0, hr, cT, sRT,
       hr, cT, sRT, 0, nullptr, 0, cR, cT, cG, BC, 0.5f, 1.f, 1, 0);
  gemm(stream, Mor, cR, 0, 0, hr, cT, sRT, 0, ho, cT, sNT,
       ho, cT, sNT, 0, nullptr, 0, cN, cT, cR, BC, 0.5f, 1.f, 1, 0);
  gemm(stream, Mor, cR, 0, 1, ho, cT, sNT, 0, hr, cT, sRT,
       out_xc, cT, sRT, 0, nullptr, 0, cR, cT, cN, BC, 0.5f, 1.f, 1, 0);
  gemm(stream, Mrg, cG, 0, 1, out_xc, cT, sRT, 0, xg, cT, sGT,
       out_xs, cT, sGT, 0, nullptr, 0, cG, cT, cR, BC, 0.5f, 1.f, 1, 0);

  // ===== STransformer main =====
  const long long sBNTE = (long long)cN * cT * cE;          // per-b (N,T,E)
  // Xc = transpose(conv1(ho)) -> (B,N,T,E); q = transpose(conv1(query))
  gemm(stream, Wc1, cC, 0, 0, ho, (int)sNT, sBNT, 0, nullptr, 0, 0,
       Xc, cE, sBNTE, 1, bc1, 1, cE, (int)sNT, cC, cB, 1.f, 0.f, 0, 0);
  gemm(stream, Wc1, cC, 0, 0, query, (int)sNT, sBNT, 0, nullptr, 0, 0,
       qb, cE, sBNTE, 1, bc1, 1, cE, (int)sNT, cC, cB, 1.f, 0.f, 0, 0);
  // ds = D_S @ Wemb^T + bemb  (400 x 64)
  gemm(stream, D_S, cN, 0, 0, Wemb, cN, 0, 1, nullptr, 0, 0,
       dsb, cE, 0, 0, bemb, 2, cN, cE, cN, 1, 1.f, 0.f, 0, 0);
  q_add_ds_kernel<<<2048, 256, 0, stream>>>(qb, dsb, (long long)BNTE);

  // qq/kk/vv: per head (batch=2): (TOK x 32) @ W^T
  const int M_TOK = (int)TOK;
  gemm(stream, qb, cE, cDH, 0, Wq, cDH, 0, 1, nullptr, 0, 0,
       qqb, cE, cDH, 0, nullptr, 0, M_TOK, cDH, cDH, 2, 1.f, 0.f, 0, 0);
  gemm(stream, qb, cE, cDH, 0, Wk, cDH, 0, 1, nullptr, 0, 0,
       kkb, cE, cDH, 0, nullptr, 0, M_TOK, cDH, cDH, 2, 1.f, 0.f, 0, 0);
  gemm(stream, qb, cE, cDH, 0, Wv, cDH, 0, 1, nullptr, 0, 0,
       vvb, cE, cDH, 0, nullptr, 0, M_TOK, cDH, cDH, 2, 1.f, 0.f, 0, 0);

  // fused attention
  {
    dim3 grid((cN + 15) / 16, cT * 2, cB);                  // (25, 24, 16)
    attn_kernel<<<grid, 32, 0, stream>>>(qqb, kkb, vvb, ato);
  }
  // attn fc: (TOK x 64) @ W_fc^T + b_fc
  gemm(stream, ato, cE, 0, 0, Wfc, cE, 0, 1, nullptr, 0, 0,
       afc, cE, 0, 0, bfc, 2, M_TOK, cE, cE, 1, 1.f, 0.f, 0, 0);
  // x = LN(attn + q)
  ln_residual_kernel<<<(unsigned)TOK, 64, 0, stream>>>(afc, qb, ln1w, ln1b, xln);
  // ff = relu(x @ Wff1^T + b1) @ Wff2^T + b2    (ff2 reuses `ato`)
  gemm(stream, xln, cE, 0, 0, Wff1, cE, 0, 1, nullptr, 0, 0,
       ff1, cFE, 0, 0, bff1, 2, M_TOK, cFE, cE, 1, 1.f, 0.f, 1, 0);
  gemm(stream, ff1, cFE, 0, 0, Wff2, cFE, 0, 1, nullptr, 0, 0,
       ato, cE, 0, 0, bff2, 2, M_TOK, cE, cFE, 1, 1.f, 0.f, 0, 0);
  // U_S = LN(ff + x)   (reuses `afc`)
  ln_residual_kernel<<<(unsigned)TOK, 64, 0, stream>>>(ato, xln, ln2w, ln2b, afc);
  // g = sigmoid(U_S@Wfs^T + bfs + Xc@Wfg^T + bfg)   (reuses `qqb`)
  gemm(stream, afc, cE, 0, 0, Wfs, cE, 0, 1, nullptr, 0, 0,
       qqb, cE, 0, 0, bfs, 2, M_TOK, cE, cE, 1, 1.f, 0.f, 0, 0);
  gemm(stream, Xc, cE, 0, 0, Wfg, cE, 0, 1, qqb, cE, 0,
       qqb, cE, 0, 0, bfg, 2, M_TOK, cE, cE, 1, 1.f, 1.f, 0, 2);
  // gated = g*U_S + (1-g)*Xc   (reuses `kkb`)
  gate_mix_kernel<<<2048, 256, 0, stream>>>(qqb, afc, Xc, kkb, (long long)BNTE);
  // final: out[b] (32 x 4800) = Wc11 @ gated_b^T + bc11  -> d_out main slice
  gemm(stream, Wc11, cE, 0, 0, kkb, cE, sBNTE, 1, nullptr, 0, 0,
       out_main, (int)sNT, sBNT, 0, bc11, 1, cC, (int)sNT, cE, cB, 1.f, 0.f, 0, 0);
}
